// DistMultUnit_42855183679604
// MI455X (gfx1250) — compile-verified, matrix operand order checked by reading, not run on hardware
//
#include <hip/hip_runtime.h>
#include <hip/hip_bf16.h>

// DistMult scoring on MI455X (gfx1250):
//   scores[b, e] = - sum_d (Eent[h[b],d] * Rrel[r[b],d]) * Eent[e,d]
// Q = -(h*r) in bf16 (K padded 200->224, negation folded in), GEMM via
// v_wmma_f32_16x16x32_bf16. Each wave computes ALL 256 batch rows x 16 entity
// columns so the 400MB entity matrix streams from HBM exactly once (NT loads);
// Q lives in LDS (114KB of the 320KB/WGP). A-fragment LDS loads are pipelined
// 3 WMMAs ahead; two *opaque* base pointers keep every ds_load offset inside
// the 16-bit DS immediate so no in-loop VALU address math (no WMMA WAR NOPs).

typedef __attribute__((ext_vector_type(16))) __bf16 v16bf;
typedef __attribute__((ext_vector_type(8)))  __bf16 v8bf;
typedef __attribute__((ext_vector_type(8)))  float  v8f;
typedef __attribute__((ext_vector_type(4)))  float  v4f;

#define NUM_E  500000
#define DIM    200
#define KPAD   224          // 7 * 32
#define NKSTEP 7
#define BATCH  256
#define NPW    16           // entity columns per wave
#define WAVES  8            // waves per block
#define NPB    (NPW*WAVES)  // 128 entity columns per block
#define NSTEPS (NKSTEP*16)  // 112 wmma per wave

// ---------------- Kernel 1: Q[b, k] = bf16( -E[h[b],k] * R[r[b],k] ), padded to KPAD
__global__ void __launch_bounds__(KPAD) build_q_kernel(
    const int* __restrict__ x, const float* __restrict__ ent,
    const float* __restrict__ rel, __bf16* __restrict__ q) {
  const int b = blockIdx.x;        // 0..255
  const int d = threadIdx.x;       // 0..223
  const int h = x[2 * b + 0];
  const int r = x[2 * b + 1];
  float v = 0.0f;
  if (d < DIM) {
    v = -(ent[(long long)h * DIM + d] * rel[r * DIM + d]);  // fold output negation
  }
  q[b * KPAD + d] = (__bf16)v;
}

// B fragment (32x16 bf16, CDNA5 16-bit B layout) for this lane from 16 f32 of
// one entity row: lanes 0-15 take K {k..k+7, k+16..k+23} with k = kk;
// lanes 16-31 the same pattern with k = kk+8. NT loads: E is streamed once.
__device__ __forceinline__ v16bf make_bfrag_full(const float* __restrict__ erow, int kbase) {
  const v4f* p0 = (const v4f*)(erow + kbase);        // kbase mult of 8 -> aligned
  const v4f* p1 = (const v4f*)(erow + kbase + 16);
  v4f f0 = __builtin_nontemporal_load(p0);
  v4f f1 = __builtin_nontemporal_load(p0 + 1);
  v4f f2 = __builtin_nontemporal_load(p1);
  v4f f3 = __builtin_nontemporal_load(p1 + 1);
  v16bf b;
  b[0]  = (__bf16)f0[0]; b[1]  = (__bf16)f0[1]; b[2]  = (__bf16)f0[2]; b[3]  = (__bf16)f0[3];
  b[4]  = (__bf16)f1[0]; b[5]  = (__bf16)f1[1]; b[6]  = (__bf16)f1[2]; b[7]  = (__bf16)f1[3];
  b[8]  = (__bf16)f2[0]; b[9]  = (__bf16)f2[1]; b[10] = (__bf16)f2[2]; b[11] = (__bf16)f2[3];
  b[12] = (__bf16)f3[0]; b[13] = (__bf16)f3[1]; b[14] = (__bf16)f3[2]; b[15] = (__bf16)f3[3];
  return b;
}

__device__ __forceinline__ v16bf make_bfrag_tail(const float* __restrict__ erow, int kbase) {
  v16bf b;
#pragma unroll
  for (int i = 0; i < 8; ++i) {
    int k0 = kbase + i;
    int k1 = kbase + 16 + i;
    b[i]     = (__bf16)((k0 < DIM) ? __builtin_nontemporal_load(erow + k0) : 0.0f);
    b[8 + i] = (__bf16)((k1 < DIM) ? __builtin_nontemporal_load(erow + k1) : 0.0f);
  }
  return b;
}

// ---------------- Kernel 2: scores = (Qneg x E^T) via WMMA bf16 -> f32
__global__ void __launch_bounds__(256) distmult_gemm_kernel(
    const float* __restrict__ E, const __bf16* __restrict__ Q,
    float* __restrict__ out) {
  extern __shared__ __bf16 qlds[];   // BATCH * KPAD bf16 = 114688 bytes

  const int wave = threadIdx.x >> 5;
  const int lane = threadIdx.x & 31;
  const long long n0 = ((long long)blockIdx.x * WAVES + wave) * NPW;
  const bool active = (n0 < NUM_E);  // wave-uniform

  const int half = lane >> 4;          // lane group 0 or 1
  const int l    = lane & 15;
  const int n    = (int)(active ? n0 : 0) + l;   // clamp to keep loads in-bounds
  const int koff = half * 8;           // K sub-offset per 16-bit A/B VGPR layout

  const float* erow = E + (long long)n * DIM;

  // --- Preload all 7 B fragments BEFORE the barrier: global latency overlaps
  //     the LDS staging of Q. 24x global_load_b128 (NT) + guarded tail + cvt.
  v16bf bf[NKSTEP];
  if (active) {
#pragma unroll
    for (int s = 0; s < NKSTEP - 1; ++s) bf[s] = make_bfrag_full(erow, s * 32 + koff);
    bf[NKSTEP - 1] = make_bfrag_tail(erow, (NKSTEP - 1) * 32 + koff);
  }

  // --- Stage all of Q into LDS (coalesced 16B chunks; Q is L2-hot) ---
  {
    const v4f* src = (const v4f*)Q;
    v4f* dst = (v4f*)qlds;
    const int total = (BATCH * KPAD * 2) / 16;   // 7168 chunks
    for (int i = threadIdx.x; i < total; i += blockDim.x) dst[i] = src[i];
  }
  __syncthreads();
  if (!active) return;

  v8f zero = {};
  v8f acc[16];
#pragma unroll
  for (int mt = 0; mt < 16; ++mt) acc[mt] = zero;

  // Two A base pointers so every ds_load uses a static immediate offset
  // (16-bit DS offset limit is 65535; mt*7168 alone reaches 107520):
  //   mt 0..8  from abase0 (max offset  8*7168 + 6*64 + 32 = 57760)
  //   mt 9..15 from abase1 (max offset  6*7168 + 6*64 + 32 = 43424)
  // The split offset is made OPAQUE so the optimizer cannot re-associate
  // abase1 back into abase0+const (which forced in-loop v_add address math
  // into WMMA source registers -> mandatory 4x v_nop WAR hazard gaps).
  const __bf16* abase0 = qlds + l * KPAD + koff;
  int split_elems = 9 * 16 * KPAD;              // element offset of mt==9 row
  asm("" : "+v"(split_elems));                  // opaque to the optimizer
  const __bf16* abase1 = abase0 + split_elems;

  // 4-slot rotating A buffer; fragment for step t is loaded at step t-3 so the
  // ds_load latency hides behind ~3 WMMAs.
  v8bf alo[4], ahi[4];
#define LOAD_A(t, slot)                                                        \
  {                                                                            \
    const int kk_ = (t) >> 4, mt_ = (t) & 15;                                  \
    const __bf16* qr_ = (mt_ < 9)                                              \
        ? (abase0 + mt_ * 16 * KPAD + kk_ * 32)                                \
        : (abase1 + (mt_ - 9) * 16 * KPAD + kk_ * 32);                         \
    alo[slot] = *(const v8bf*)(qr_);                                           \
    ahi[slot] = *(const v8bf*)(qr_ + 16);                                      \
  }

  LOAD_A(0, 0)
  LOAD_A(1, 1)
  LOAD_A(2, 2)
#pragma unroll
  for (int t = 0; t < NSTEPS; ++t) {
    if (t + 3 < NSTEPS) LOAD_A(t + 3, (t + 3) & 3)
    const int kk = t >> 4;
    const int mt = t & 15;
    v16bf av = __builtin_shufflevector(alo[t & 3], ahi[t & 3],
                                       0, 1, 2, 3, 4, 5, 6, 7,
                                       8, 9, 10, 11, 12, 13, 14, 15);
    acc[mt] = __builtin_amdgcn_wmma_f32_16x16x32_bf16(
        /*neg_a=*/false, av, /*neg_b=*/false, bf[kk],
        /*c_mod=*/(short)0, acc[mt], /*reuse_a=*/false, /*reuse_b=*/false);
  }
#undef LOAD_A

  // --- Store (NT: scores never re-read). C/D layout: VGPR i -> M = i
  //     (lanes 0-15) / M = i+8 (lanes 16-31); negation already in Q. ---
#pragma unroll
  for (int mt = 0; mt < 16; ++mt) {
#pragma unroll
    for (int i = 0; i < 8; ++i) {
      const int row = mt * 16 + i + half * 8;              // batch row
      __builtin_nontemporal_store(acc[mt][i], &out[row * NUM_E + n]);
    }
  }
}

extern "C" void kernel_launch(void* const* d_in, const int* in_sizes, int n_in,
                              void* d_out, int out_size, void* d_ws, size_t ws_size,
                              hipStream_t stream) {
  const int*   x   = (const int*)d_in[0];      // (256,2) indices
  const float* ent = (const float*)d_in[1];    // (500000,200) f32
  const float* rel = (const float*)d_in[2];    // (1000,200) f32
  float*       out = (float*)d_out;            // (256,500000) f32

  __bf16* q = (__bf16*)d_ws;                   // 256*224 bf16 = 114688 bytes

  // Kernel 1: build padded, negated bf16 Q
  build_q_kernel<<<BATCH, KPAD, 0, stream>>>(x, ent, rel, q);

  // Kernel 2: WMMA GEMM, dynamic LDS holds all of Q
  const int grid = (NUM_E + NPB - 1) / NPB;    // 3907 blocks
  const size_t lds_bytes = (size_t)BATCH * KPAD * sizeof(__bf16);  // 114688
  distmult_gemm_kernel<<<grid, 256, lds_bytes, stream>>>(ent, q, out);
}
